// GAT2_48593259987028
// MI455X (gfx1250) — matrix operand order repaired and checked by reading, not compiled, not run on hardware
//
#include <hip/hip_runtime.h>
#include <math.h>

// ---------------- problem constants (match reference) ----------------
#define N_NODES   8000
#define N_EDGES   96000
#define E_TOT     (N_EDGES + N_NODES)   // self-loops appended
#define N_FEAT    256
#define N_GRAPHS  64
#define N_CLASSES 10
#define HEADS     8
#define C1        64                    // conv1 per-head channels (concat -> 512)
#define C2        128                   // conv2 per-head channels (mean -> 128)
#define D1        (HEADS * C1)          // 512
#define D2        (HEADS * C2)          // 1024
#define NEG_SLOPE 0.2f

typedef __attribute__((ext_vector_type(16))) _Float16 v16h;
typedef __attribute__((ext_vector_type(4)))  _Float16 v4h;
typedef __attribute__((ext_vector_type(8)))  float    v8f;

// float atomic max via signed/unsigned int ordering trick.
// Requires memory initialized with a float written normally (-inf here).
__device__ __forceinline__ void atomicMaxF(float* addr, float val) {
    if (val >= 0.0f)
        atomicMax((int*)addr, __float_as_int(val));
    else
        atomicMin((unsigned int*)addr, (unsigned int)__float_as_int(val));
}

// ---------------- utility ----------------
__global__ void fill_f32(float* __restrict__ p, float v, int n) {
    int i = blockIdx.x * blockDim.x + threadIdx.x;
    if (i < n) p[i] = v;
}

// ---------------- WMMA GEMM: C[M,N] = A[M,K] @ B[K,N] ----------------
// fp32 in memory, f16 multiply / f32 accumulate (v_wmma_f32_16x16x32_f16).
// Block = 8 wave32 = macro-tile 128(M) x 32(N). B tile staged through LDS
// (f16, column-major, 80B column stride: 16B aligned for ds_load_b128 and
// conflict-free fragment reads). Each wave reuses its A fragment for two
// WMMAs. M%16==0, N%32==0, K%32==0. Fragment layouts per CDNA5 ISA 7.12.2.
#define GEMM_WAVES 8
#define BPAD 40   // halfs per staged B column (32 data + 8 pad) = 80 bytes

__global__ void __launch_bounds__(256)
gemm_wmma_f16(const float* __restrict__ A,
              const float* __restrict__ B,
              float* __restrict__ C,
              int M, int N, int K) {
    __shared__ _Float16 lbs[32 * BPAD];     // [col][k], 2560 B

    const int lane   = threadIdx.x & 31;
    const int wave   = threadIdx.x >> 5;
    const int tilesM = M >> 4;

    int tmIdx = blockIdx.y * GEMM_WAVES + wave;
    const bool active = (tmIdx < tilesM);   // wave-uniform
    if (!active) tmIdx = tilesM - 1;        // clamp: redundant compute, no store
    const int tm  = tmIdx << 4;
    const int tn  = blockIdx.x << 5;        // 32 output columns per block
    const int row = lane & 15;              // M-row for A, N-col for B/C/D
    const int kh  = lane >> 4;              // which K-half this lane holds

    // staging assignment: this thread owns (scol, sk..sk+3) of the 32x32 B tile
    const int scol = threadIdx.x & 31;
    const int sk   = (threadIdx.x >> 5) * 4;

    v8f acc0 = {}, acc1 = {};
    const float* Arow = A + (size_t)(tm + row) * K;

    for (int kb = 0; kb < K; kb += 32) {
        __syncthreads();                    // previous reads done before restage
        // cooperative stage: B[kb..kb+31][tn..tn+31] -> LDS f16 col-major.
        // 4 coalesced b32 rows per thread -> one packed ds_store_b64.
        {
            const float* bp = B + (size_t)(kb + sk) * N + tn + scol;
            float f0 = bp[0];
            float f1 = bp[(size_t)N];
            float f2 = bp[2 * (size_t)N];
            float f3 = bp[3 * (size_t)N];
            v4h pk;
            pk[0] = (_Float16)f0;
            pk[1] = (_Float16)f1;
            pk[2] = (_Float16)f2;
            pk[3] = (_Float16)f3;
            *(v4h*)(&lbs[scol * BPAD + sk]) = pk;   // 8B aligned
        }
        __syncthreads();

        // A 16x32 fragment: lane<16 holds K {0..7,16..23}; lane>=16 {8..15,24..31}
        v16h a;
#pragma unroll
        for (int i = 0; i < 8; ++i) {
            a[i]     = (_Float16)Arow[kb + kh * 8 + i];
            a[i + 8] = (_Float16)Arow[kb + 16 + kh * 8 + i];
        }
        if (kb + 32 < K)
            __builtin_prefetch(Arow + kb + 32, 0, 1);   // global_prefetch_b8

        // B 32x16 fragments from LDS: lane<16 K {0..7,16..23}; lane>=16 {8..15,24..31}
        v16h b0, b1;
#pragma unroll
        for (int j = 0; j < 8; ++j) {
            b0[j]     = lbs[row * BPAD + kh * 8 + j];
            b0[j + 8] = lbs[row * BPAD + 16 + kh * 8 + j];
            b1[j]     = lbs[(row + 16) * BPAD + kh * 8 + j];
            b1[j + 8] = lbs[(row + 16) * BPAD + 16 + kh * 8 + j];
        }
        acc0 = __builtin_amdgcn_wmma_f32_16x16x32_f16(
            false, a, false, b0, (short)0, acc0, false, false);
        acc1 = __builtin_amdgcn_wmma_f32_16x16x32_f16(
            false, a, false, b1, (short)0, acc1, false, false);
    }

    if (active) {
        // D 16x16 f32: element r -> M = r + 8*kh, N = lane&15
#pragma unroll
        for (int r = 0; r < 8; ++r) {
            C[(size_t)(tm + r + kh * 8) * N + tn + row]      = acc0[r];
            C[(size_t)(tm + r + kh * 8) * N + tn + 16 + row] = acc1[r];
        }
    }
}

// ---------------- edge kernels (templated on per-head channels) ----------------
// xl/xr layout: [N, HEADS, C] row-major. logit: [E_TOT, HEADS]. m,s: [N, HEADS].
template <int C>
__global__ void edge_logits_kernel(const float* __restrict__ xl,
                                   const float* __restrict__ xr,
                                   const float* __restrict__ att,
                                   const int*  __restrict__ esrc,
                                   const int*  __restrict__ edst,
                                   float* __restrict__ logit,
                                   float* __restrict__ m) {
    int t = blockIdx.x * blockDim.x + threadIdx.x;
    if (t >= E_TOT * HEADS) return;
    int e = t / HEADS, h = t % HEADS;
    int sn = (e < N_EDGES) ? esrc[e] : (e - N_EDGES);
    int dn = (e < N_EDGES) ? edst[e] : (e - N_EDGES);
    const float4* xls = (const float4*)(xl + ((size_t)sn * HEADS + h) * C);
    const float4* xrd = (const float4*)(xr + ((size_t)dn * HEADS + h) * C);
    const float4* av  = (const float4*)(att + h * C);
    float acc = 0.0f;
#pragma unroll 4
    for (int q = 0; q < C / 4; ++q) {
        float4 l = xls[q], r = xrd[q], w = av[q];
        float v;
        v = l.x + r.x; v = (v > 0.0f) ? v : NEG_SLOPE * v; acc += v * w.x;
        v = l.y + r.y; v = (v > 0.0f) ? v : NEG_SLOPE * v; acc += v * w.y;
        v = l.z + r.z; v = (v > 0.0f) ? v : NEG_SLOPE * v; acc += v * w.z;
        v = l.w + r.w; v = (v > 0.0f) ? v : NEG_SLOPE * v; acc += v * w.w;
    }
    logit[t] = acc;
    atomicMaxF(&m[dn * HEADS + h], acc);
}

__global__ void edge_expsum_kernel(float* __restrict__ logit,
                                   const float* __restrict__ m,
                                   const int* __restrict__ edst,
                                   float* __restrict__ s) {
    int t = blockIdx.x * blockDim.x + threadIdx.x;
    if (t >= E_TOT * HEADS) return;
    int e = t / HEADS, h = t % HEADS;
    int dn = (e < N_EDGES) ? edst[e] : (e - N_EDGES);
    float a = expf(logit[t] - m[dn * HEADS + h]);
    logit[t] = a;                              // keep unnormalized alpha
    atomicAdd(&s[dn * HEADS + h], a);
}

template <int C>
__global__ void edge_scatter_kernel(const float* __restrict__ xl,
                                    const float* __restrict__ alpha,  // = logit buf
                                    const float* __restrict__ s,
                                    const int* __restrict__ esrc,
                                    const int* __restrict__ edst,
                                    float* __restrict__ out) {
    int t = blockIdx.x * blockDim.x + threadIdx.x;
    if (t >= E_TOT * HEADS) return;
    int e = t / HEADS, h = t % HEADS;
    int sn = (e < N_EDGES) ? esrc[e] : (e - N_EDGES);
    int dn = (e < N_EDGES) ? edst[e] : (e - N_EDGES);
    float coef = alpha[t] / s[dn * HEADS + h];
    const float4* xls = (const float4*)(xl + ((size_t)sn * HEADS + h) * C);
    float* od = out + ((size_t)dn * HEADS + h) * C;
#pragma unroll 4
    for (int q = 0; q < C / 4; ++q) {
        float4 v = xls[q];                     // b128 gather (L2-resident)
        atomicAdd(&od[q * 4 + 0], v.x * coef);
        atomicAdd(&od[q * 4 + 1], v.y * coef);
        atomicAdd(&od[q * 4 + 2], v.z * coef);
        atomicAdd(&od[q * 4 + 3], v.w * coef);
    }
}

// h1 = elu(out1 + b1), in place over [N, D1]
__global__ void finish_conv1_kernel(float* __restrict__ out1,
                                    const float* __restrict__ b1) {
    int t = blockIdx.x * blockDim.x + threadIdx.x;
    if (t >= N_NODES * D1) return;
    float v = out1[t] + b1[t % D1];
    out1[t] = (v > 0.0f) ? v : (expf(v) - 1.0f);
}

// h2[n,c] = mean_h(out2[n,h,c]) + b2[c]
__global__ void finish_conv2_kernel(const float* __restrict__ out2,
                                    const float* __restrict__ b2,
                                    float* __restrict__ h2) {
    int t = blockIdx.x * blockDim.x + threadIdx.x;
    if (t >= N_NODES * C2) return;
    int n = t / C2, c = t % C2;
    float acc = 0.0f;
#pragma unroll
    for (int h = 0; h < HEADS; ++h)
        acc += out2[((size_t)n * HEADS + h) * C2 + c];
    h2[t] = acc * (1.0f / HEADS) + b2[c];
}

__global__ void pool_kernel(const float* __restrict__ h2,
                            const int* __restrict__ batch,
                            float* __restrict__ pooled,
                            float* __restrict__ cnt) {
    int t = blockIdx.x * blockDim.x + threadIdx.x;
    if (t >= N_NODES * C2) return;
    int n = t / C2, c = t % C2;
    int g = batch[n];
    atomicAdd(&pooled[g * C2 + c], h2[t]);
    if (c == 0) atomicAdd(&cnt[g], 1.0f);
}

__global__ void final_linear_kernel(const float* __restrict__ pooled,
                                    const float* __restrict__ cnt,
                                    const float* __restrict__ Wlin,
                                    const float* __restrict__ blin,
                                    float* __restrict__ out) {
    int t = blockIdx.x * blockDim.x + threadIdx.x;
    if (t >= N_GRAPHS * N_CLASSES) return;
    int g = t / N_CLASSES, k = t % N_CLASSES;
    float inv = 1.0f / fmaxf(cnt[g], 1.0f);
    float acc = blin[k];
#pragma unroll 8
    for (int c = 0; c < C2; ++c)
        acc += (pooled[g * C2 + c] * inv) * Wlin[c * N_CLASSES + k];
    out[t] = acc;
}

// ---------------- host-side orchestration ----------------
static inline int cdiv(int a, int b) { return (a + b - 1) / b; }

static inline void launch_gemm(const float* A, const float* B, float* C,
                               int M, int N, int K, hipStream_t stream) {
    dim3 grid(N / 32, cdiv(M / 16, GEMM_WAVES));
    gemm_wmma_f16<<<grid, 256, 0, stream>>>(A, B, C, M, N, K);
}

extern "C" void kernel_launch(void* const* d_in, const int* in_sizes, int n_in,
                              void* d_out, int out_size, void* d_ws, size_t ws_size,
                              hipStream_t stream) {
    (void)in_sizes; (void)n_in; (void)out_size; (void)ws_size;

    const float* x    = (const float*)d_in[0];
    const int*   eidx = (const int*)  d_in[1];   // [2, N_EDGES]
    const int*   batc = (const int*)  d_in[2];
    const float* Wl1  = (const float*)d_in[3];
    const float* Wr1  = (const float*)d_in[4];
    const float* att1 = (const float*)d_in[5];
    const float* b1   = (const float*)d_in[6];
    const float* Wl2  = (const float*)d_in[7];
    const float* Wr2  = (const float*)d_in[8];
    const float* att2 = (const float*)d_in[9];
    const float* b2   = (const float*)d_in[10];
    const float* Wlin = (const float*)d_in[11];
    const float* blin = (const float*)d_in[12];
    const int* esrc = eidx;
    const int* edst = eidx + N_EDGES;
    float* out = (float*)d_out;

    // workspace layout (floats); ~123 MB total -> fully L2-resident (192 MB).
    float* ws = (float*)d_ws;
    size_t off = 0;
    float* xl1   = ws + off; off += (size_t)N_NODES * D1;
    float* xr1   = ws + off; off += (size_t)N_NODES * D1;   // contiguous after xl1
    float* out1  = ws + off; off += (size_t)N_NODES * D1;   // accum -> h1 in place
    float* xl2   = ws + off; off += (size_t)N_NODES * D2;
    float* xr2   = ws + off; off += (size_t)N_NODES * D2;
    float* h2    = ws + off; off += (size_t)N_NODES * C2;
    float* logit = ws + off; off += (size_t)E_TOT * HEADS;
    float* mbuf  = ws + off; off += (size_t)N_NODES * HEADS;
    float* sbuf  = ws + off; off += (size_t)N_NODES * HEADS;
    float* pool  = ws + off; off += (size_t)N_GRAPHS * C2;
    float* cnt   = ws + off; off += (size_t)N_GRAPHS;
    float* out2  = xl1;      // alias xl1+xr1 (dead by the time out2 is used)

    const int TB = 256;
    const int EH = E_TOT * HEADS;

    // ---- conv1: xl1 = x @ Wl1 ; xr1 = x @ Wr1   (8000 x 256 x 512) ----
    launch_gemm(x, Wl1, xl1, N_NODES, D1, N_FEAT, stream);
    launch_gemm(x, Wr1, xr1, N_NODES, D1, N_FEAT, stream);

    fill_f32<<<cdiv(N_NODES * HEADS, TB), TB, 0, stream>>>(mbuf, -INFINITY, N_NODES * HEADS);
    fill_f32<<<cdiv(N_NODES * HEADS, TB), TB, 0, stream>>>(sbuf, 0.0f, N_NODES * HEADS);
    fill_f32<<<cdiv(N_NODES * D1, TB), TB, 0, stream>>>(out1, 0.0f, N_NODES * D1);

    edge_logits_kernel<C1><<<cdiv(EH, TB), TB, 0, stream>>>(xl1, xr1, att1, esrc, edst, logit, mbuf);
    edge_expsum_kernel<<<cdiv(EH, TB), TB, 0, stream>>>(logit, mbuf, edst, sbuf);
    edge_scatter_kernel<C1><<<cdiv(EH, TB), TB, 0, stream>>>(xl1, logit, sbuf, esrc, edst, out1);
    finish_conv1_kernel<<<cdiv(N_NODES * D1, TB), TB, 0, stream>>>(out1, b1);  // out1 == h1 now

    // ---- conv2: xl2 = h1 @ Wl2 ; xr2 = h1 @ Wr2   (8000 x 512 x 1024) ----
    launch_gemm(out1, Wl2, xl2, N_NODES, D2, D1, stream);
    launch_gemm(out1, Wr2, xr2, N_NODES, D2, D1, stream);

    fill_f32<<<cdiv(N_NODES * HEADS, TB), TB, 0, stream>>>(mbuf, -INFINITY, N_NODES * HEADS);
    fill_f32<<<cdiv(N_NODES * HEADS, TB), TB, 0, stream>>>(sbuf, 0.0f, N_NODES * HEADS);
    fill_f32<<<cdiv(N_NODES * D2, TB), TB, 0, stream>>>(out2, 0.0f, N_NODES * D2);

    edge_logits_kernel<C2><<<cdiv(EH, TB), TB, 0, stream>>>(xl2, xr2, att2, esrc, edst, logit, mbuf);
    edge_expsum_kernel<<<cdiv(EH, TB), TB, 0, stream>>>(logit, mbuf, edst, sbuf);
    edge_scatter_kernel<C2><<<cdiv(EH, TB), TB, 0, stream>>>(xl2, logit, sbuf, esrc, edst, out2);
    finish_conv2_kernel<<<cdiv(N_NODES * C2, TB), TB, 0, stream>>>(out2, b2, h2);

    // ---- global mean pool + final linear ----
    fill_f32<<<cdiv(N_GRAPHS * C2, TB), TB, 0, stream>>>(pool, 0.0f, N_GRAPHS * C2);
    fill_f32<<<1, 64, 0, stream>>>(cnt, 0.0f, N_GRAPHS);
    pool_kernel<<<cdiv(N_NODES * C2, TB), TB, 0, stream>>>(h2, batc, pool, cnt);
    final_linear_kernel<<<cdiv(N_GRAPHS * N_CLASSES, 128), 128, 0, stream>>>(pool, cnt, Wlin, blin, out);
}
